// LiquidEmbedding_60833916781076
// MI455X (gfx1250) — matrix operand order repaired and checked by reading, not compiled, not run on hardware
//
#include <hip/hip_runtime.h>
#include <cstdint>
#include <cstddef>

// ---------------- problem constants ----------------
#define DM   1024          // feature dim
#define SEQ  2048          // sequence length
#define BNUM 8             // batch
#define TOK  (BNUM*SEQ)    // 16384 tokens
#define TD   ((size_t)TOK*DM)

typedef __attribute__((ext_vector_type(16))) __bf16 v16bf;
typedef __attribute__((ext_vector_type(8)))  float  v8f;

__device__ __forceinline__ __bf16 f2bf(float f) {
  unsigned u = __builtin_bit_cast(unsigned, f);
  unsigned short s = (unsigned short)((u + 0x7FFFu + ((u >> 16) & 1u)) >> 16); // RNE
  return __builtin_bit_cast(__bf16, s);
}

__device__ __forceinline__ float gelu_erf(float v) {
  return 0.5f * v * (1.0f + erff(v * 0.70710678118654752f));
}

// ---------------- init / convert ----------------
__global__ void init_kernel(int* counts, float* accum) {
  int i = threadIdx.x;
  if (i < 16) counts[i] = 0;       // counts[8] + nkeep[8] zero
  if (i < 2)  accum[i]  = 0.0f;
}

__global__ void cvt_bf16_kernel(const float* __restrict__ in, __bf16* __restrict__ out, int n) {
  int i = blockIdx.x * 256 + threadIdx.x;
  if (i < n) out[i] = f2bf(in[i]);
}

// ---------------- embedding (writes bf16 activations) ----------------
__global__ void embed_kernel(const int* __restrict__ x, const float* __restrict__ tok,
                             const float* __restrict__ pos, __bf16* __restrict__ out) {
  size_t i = (size_t)blockIdx.x * 256 + threadIdx.x;   // i < TOK*DM
  int d = (int)(i & (DM - 1));
  size_t t = i >> 10;                                  // DM == 1024
  int token = x[t];
  int s = (int)(t & (SEQ - 1));                        // SEQ power of two
  out[i] = f2bf(tok[(size_t)token * DM + d] + pos[(size_t)s * DM + d]);
}

// ---------------- generic bf16 WMMA GEMM (double-buffered LDS) ----------------
// out[t, n] = act( sum_{k} A_conv[t, k] * W[k, n] + bias[n] )
// KW==3: conv1d k=3 SAME (A_conv[t, dk*DM+i] = A[t+dk-1, i] with per-batch zero halo)
// KW==1: plain GEMM, Ktot = DM.
// Tile: 128(M) x 128(N), 256 threads = 8 waves, each wave -> 16 rows x 128 cols.
__device__ __forceinline__ void stage_tiles(const __bf16* __restrict__ A,
                                            const __bf16* __restrict__ W,
                                            __bf16 (*As)[32], __bf16 (*Bs)[32],
                                            int N, int KW, int m0, int n0, int batch,
                                            int k0, int tid) {
  const int center = KW >> 1;
  const int dk = k0 >> 10;                   // k0 / DM
  const int i0 = k0 - (dk << 10);
#pragma unroll
  for (int c = tid; c < 512; c += 256) {     // A: 128 rows x 4 chunks of 8 bf16
    int row  = c >> 2;
    int ioff = (c & 3) << 3;
    int t    = m0 + row;
    int srcs = (t - batch * SEQ) + dk - center;
    uint4 val = make_uint4(0u, 0u, 0u, 0u);
    if (srcs >= 0 && srcs < SEQ)
      val = *(const uint4*)(A + ((size_t)(batch * SEQ + srcs) * DM + i0 + ioff));
    *(uint4*)(&As[row][ioff]) = val;
  }
#pragma unroll
  for (int c = tid; c < 512; c += 256) {     // B: 32 k x 16 chunks of 8 n, transpose -> [n][k]
    int k   = c >> 4;
    int nch = (c & 15) << 3;
    union { uint4 q; __bf16 h[8]; } u;
    u.q = *(const uint4*)(W + ((size_t)(k0 + k) * N + n0 + nch));
#pragma unroll
    for (int j = 0; j < 8; ++j) Bs[nch + j][k] = u.h[j];
  }
}

__global__ void gemm_bf16_kernel(const __bf16* __restrict__ A, const __bf16* __restrict__ W,
                                 const float* __restrict__ bias,
                                 float* __restrict__ outF, __bf16* __restrict__ outB,
                                 int N, int KW, int doGelu) {
  __shared__ alignas(16) __bf16 As[2][128][32];   // [buf][m][k]
  __shared__ alignas(16) __bf16 Bs[2][128][32];   // [buf][n][k]

  const int tid  = threadIdx.x;
  const int lane = tid & 31;
  const int wv   = tid >> 5;
  const int ntiles = N >> 7;
  const int tm = blockIdx.x / ntiles;
  const int tn = blockIdx.x - tm * ntiles;
  const int m0 = tm << 7;
  const int n0 = tn << 7;
  const int batch = m0 / SEQ;                  // 128-token tile never crosses a batch
  const int nsteps = (KW * DM) >> 5;

  v8f acc[8];
#pragma unroll
  for (int i = 0; i < 8; ++i) acc[i] = v8f{0.f,0.f,0.f,0.f,0.f,0.f,0.f,0.f};

  // prologue: stage K-block 0 into buffer 0
  stage_tiles(A, W, As[0], Bs[0], N, KW, m0, n0, batch, 0, tid);
  __syncthreads();

  const int mrow = (wv << 4) + (lane & 15);
  const int kg   = (lane >> 4) << 3;           // A frag: 0 or 8
  const int kb   = (lane >> 4) << 4;           // unused; B frag also uses kg

  for (int step = 0; step < nsteps; ++step) {
    const int cur = step & 1;
    // stage next K-block into the other buffer while this one is consumed
    if (step + 1 < nsteps) {
      if (tid == 0)
        __builtin_prefetch(W + ((size_t)(step + 2) * 32 * N + n0), 0, 1); // global_prefetch_b8
      stage_tiles(A, W, As[cur ^ 1], Bs[cur ^ 1], N, KW, m0, n0, batch, (step + 1) << 5, tid);
    }

    union UF { v16bf v; uint4 q[2]; };
    // A fragment (ISA 16-bit A 16x32 layout)
    UF a;
    a.q[0] = *(const uint4*)(&As[cur][mrow][kg]);       // K = kg..kg+7
    a.q[1] = *(const uint4*)(&As[cur][mrow][16 + kg]);  // K = 16+kg..
    // software-pipelined B fragments: loads for tile nt+1 issue before WMMA nt,
    // so each WMMA waits on loads issued one matrix-op earlier.
    UF bcur;
    {
      const __bf16* p = &Bs[cur][lane & 15][0];
      bcur.q[0] = *(const uint4*)(p + kg);
      bcur.q[1] = *(const uint4*)(p + 16 + kg);
    }
#pragma unroll
    for (int nt = 0; nt < 8; ++nt) {
      UF bnext;
      if (nt < 7) {
        const __bf16* p = &Bs[cur][((nt + 1) << 4) + (lane & 15)][0];
        bnext.q[0] = *(const uint4*)(p + kg);
        bnext.q[1] = *(const uint4*)(p + 16 + kg);
      }
      acc[nt] = __builtin_amdgcn_wmma_f32_16x16x32_bf16(
          false, a.v, false, bcur.v, (short)0, acc[nt], false, false);
      if (nt < 7) bcur = bnext;
    }
    __syncthreads();   // next buffer fully staged; this buffer free for step+2
  }

  // ---- epilogue: bias + (optional) GELU, f32 and/or bf16 stores ----
  const int lc = lane & 15;
  const int lh = lane >> 4;
#pragma unroll
  for (int nt = 0; nt < 8; ++nt) {
    int col = n0 + (nt << 4) + lc;
    float bv = bias[col];
#pragma unroll
    for (int r = 0; r < 8; ++r) {
      int row = m0 + (wv << 4) + r + (lh << 3);
      float v = acc[nt][r] + bv;
      if (doGelu) v = gelu_erf(v);
      size_t idx = (size_t)row * N + col;
      if (outF) outF[idx] = v;
      if (outB) outB[idx] = f2bf(v);
    }
  }
}

// ---------------- layernorm over D (one block per token) ----------------
__global__ void ln_kernel(const float* __restrict__ in, const float* __restrict__ g,
                          const float* __restrict__ bta, float* __restrict__ outF,
                          __bf16* __restrict__ outB) {
  int t = blockIdx.x, tid = threadIdx.x;
  __shared__ float s1[256], s2[256];
  const float* row = in + (size_t)t * DM;
  float vals[4], a0 = 0.f, a1 = 0.f;
#pragma unroll
  for (int k = 0; k < 4; ++k) { float v = row[tid + (k << 8)]; vals[k] = v; a0 += v; a1 += v * v; }
  s1[tid] = a0; s2[tid] = a1; __syncthreads();
  for (int off = 128; off > 0; off >>= 1) {
    if (tid < off) { s1[tid] += s1[tid + off]; s2[tid] += s2[tid + off]; }
    __syncthreads();
  }
  float mean = s1[0] * (1.0f / DM);
  float var  = s2[0] * (1.0f / DM) - mean * mean;
  float rs   = rsqrtf(var + 1e-5f);
#pragma unroll
  for (int k = 0; k < 4; ++k) {
    int d = tid + (k << 8);
    float y = (vals[k] - mean) * rs * g[d] + bta[d];
    size_t idx = (size_t)t * DM + d;
    if (outF) outF[idx] = y;
    if (outB) outB[idx] = f2bf(y);
  }
}

// ---------------- Stockham radix-2 FFT, N=2048, in LDS ----------------
__device__ int fft2048(float2 buf[2][2048], int tid) {
  int cur = 0, nn = 2048, ls = 0;
  while (nn > 1) {
    int m = nn >> 1;
    int s = 1 << ls;
    for (int t = tid; t < 1024; t += 256) {
      int p = t >> ls;
      int q = t & (s - 1);
      float ang = -6.28318530717958647f * (float)p / (float)nn;
      float sn, cs; __sincosf(ang, &sn, &cs);
      float2 a  = buf[cur][q + s * p];
      float2 bb = buf[cur][q + s * (p + m)];
      float ax = a.x - bb.x, ay = a.y - bb.y;
      buf[cur ^ 1][q + 2 * s * p]     = make_float2(a.x + bb.x, a.y + bb.y);
      buf[cur ^ 1][q + 2 * s * p + s] = make_float2(ax * cs - ay * sn, ax * sn + ay * cs);
    }
    __syncthreads();
    cur ^= 1; nn >>= 1; ++ls;
  }
  return cur;
}

// forward FFT of one (b,d) line + magnitude stats; X stored line-contiguous [B*D][S]
__global__ void fft_fwd_kernel(const float* __restrict__ hn, float2* __restrict__ Xc,
                               int* __restrict__ counts) {
  int line = blockIdx.x;                 // b*DM + d
  int b = line >> 10, d = line & (DM - 1);
  int tid = threadIdx.x;
  __shared__ float2 buf[2][2048];
  __shared__ float  red[256];
#pragma unroll
  for (int k = 0; k < 8; ++k) {
    int s = tid + (k << 8);
    buf[0][s] = make_float2(hn[((size_t)(b * SEQ + s)) * DM + d], 0.f);
  }
  __syncthreads();
  int cur = fft2048(buf, tid);
  float mag[8], mx = 0.f;
#pragma unroll
  for (int k = 0; k < 8; ++k) {
    int s = tid + (k << 8);
    float2 v = buf[cur][s];
    mag[k] = sqrtf(v.x * v.x + v.y * v.y);
    mx = fmaxf(mx, mag[k]);
  }
  red[tid] = mx; __syncthreads();
  for (int off = 128; off > 0; off >>= 1) {
    if (tid < off) red[tid] = fmaxf(red[tid], red[tid + off]);
    __syncthreads();
  }
  float thr = 0.1f * red[0];
  __syncthreads();
  int cnt = 0;
#pragma unroll
  for (int k = 0; k < 8; ++k) cnt += (mag[k] > thr) ? 1 : 0;
  red[tid] = (float)cnt; __syncthreads();
  for (int off = 128; off > 0; off >>= 1) {
    if (tid < off) red[tid] += red[tid + off];
    __syncthreads();
  }
  if (tid == 0) atomicAdd(&counts[b], (int)red[0]);
#pragma unroll
  for (int k = 0; k < 8; ++k) {
    int s = tid + (k << 8);
    Xc[(size_t)line * SEQ + s] = buf[cur][s];
  }
}

__global__ void ncalc_kernel(const int* __restrict__ counts, int* __restrict__ nkeep) {
  int b = threadIdx.x;
  if (b < BNUM) {
    float c  = (float)counts[b] / (float)(SEQ * DM);
    float cr = fminf(fmaxf(0.5f * (1.0f - c), 0.1f), 1.0f);
    int Nn = (int)(cr * (float)SEQ);           // trunc, matches astype(int32)
    Nn = Nn < 1 ? 1 : (Nn > SEQ ? SEQ : Nn);
    nkeep[b] = Nn;
  }
}

// masked inverse FFT: Re(ifft(X*mask)) via conj-trick; writes bf16 for proj GEMM
__global__ void ifft_kernel(const float2* __restrict__ Xc, const int* __restrict__ nkeep,
                            __bf16* __restrict__ outB) {
  int line = blockIdx.x;
  int b = line >> 10, d = line & (DM - 1);
  int tid = threadIdx.x;
  int Nk = nkeep[b];
  __shared__ float2 buf[2][2048];
#pragma unroll
  for (int k = 0; k < 8; ++k) {
    int s = tid + (k << 8);
    float2 v = (s < Nk) ? Xc[(size_t)line * SEQ + s] : make_float2(0.f, 0.f);
    buf[0][s] = make_float2(v.x, -v.y);        // conj
  }
  __syncthreads();
  int cur = fft2048(buf, tid);
#pragma unroll
  for (int k = 0; k < 8; ++k) {
    int s = tid + (k << 8);
    outB[((size_t)(b * SEQ + s)) * DM + d] = f2bf(buf[cur][s].x * (1.0f / 2048.0f));
  }
}

// ---------------- loss reductions (one block per token) ----------------
__global__ void loss_kernel(const float* __restrict__ xo, const float* __restrict__ po,
                            const float* __restrict__ px, const float* __restrict__ pt,
                            const int* __restrict__ nkeep, float* __restrict__ accum) {
  int t = blockIdx.x;
  int b = t >> 11, s = t & (SEQ - 1);
  int tid = threadIdx.x;
  __shared__ float r[7][256];
  const float* X = xo + (size_t)t * DM;
  const float* P = po + (size_t)t * DM;
  const float* U = px + (size_t)t * DM;
  const float* V = pt + (size_t)t * DM;
  float dxp=0, nx=0, np=0, dpp=0, npx=0, npt=0, l1=0;
#pragma unroll
  for (int k = 0; k < 4; ++k) {
    int d = tid + (k << 8);
    float a = X[d], c = P[d];
    dxp += a * c; nx += a * a; np += c * c;
    float dd = a - c, ad = fabsf(dd);
    l1 += (ad < 1.0f) ? 0.5f * dd * dd : ad - 0.5f;
    float u = U[d], v = V[d];
    dpp += u * v; npx += u * u; npt += v * v;
  }
  r[0][tid]=dxp; r[1][tid]=nx; r[2][tid]=np; r[3][tid]=dpp; r[4][tid]=npx; r[5][tid]=npt; r[6][tid]=l1;
  __syncthreads();
  for (int off = 128; off > 0; off >>= 1) {
    if (tid < off) {
#pragma unroll
      for (int j = 0; j < 7; ++j) r[j][tid] += r[j][tid + off];
    }
    __syncthreads();
  }
  if (tid == 0) {
    float cs = r[0][0] / (fmaxf(sqrtf(r[1][0]), 1e-12f) * fmaxf(sqrtf(r[2][0]), 1e-12f));
    cs = fminf(fmaxf(cs, -1.0f + 1e-8f), 1.0f - 1e-8f);
    atomicAdd(&accum[0], 1.0f - cs);                       // sum over ALL tokens
    if (s < nkeep[b]) {
      float cpp = r[3][0] / (fmaxf(sqrtf(r[4][0]), 1e-12f) * fmaxf(sqrtf(r[5][0]), 1e-12f));
      float sem = (2.0f - 2.0f * cpp) * (1.0f / DM);       // mean over D of (â-b̂)²
      float l1m = r[6][0] * (1.0f / DM);
      atomicAdd(&accum[1], 0.2f * l1m + 0.2f * sem);       // masked sum
    }
  }
}

__global__ void final_kernel(const float* __restrict__ accum, const int* __restrict__ nkeep,
                             float* __restrict__ out) {
  float M = 0.f;
  for (int b = 0; b < BNUM; ++b) M += (float)nkeep[b];
  float meanCos = accum[0] / (float)TOK;
  *out = (0.6f * meanCos * M + accum[1]) / (M + 1e-8f);
}

// ---------------- host launcher ----------------
extern "C" void kernel_launch(void* const* d_in, const int* in_sizes, int n_in,
                              void* d_out, int out_size, void* d_ws, size_t ws_size,
                              hipStream_t stream) {
  (void)in_sizes; (void)n_in; (void)out_size; (void)ws_size;

  const int*   x        = (const int*)  d_in[0];
  const float* tok_emb  = (const float*)d_in[1];
  const float* pos_emb  = (const float*)d_in[2];
  const float* conv1_w  = (const float*)d_in[3];
  const float* conv1_b  = (const float*)d_in[4];
  const float* conv2_w  = (const float*)d_in[5];
  const float* conv2_b  = (const float*)d_in[6];
  const float* proj_w   = (const float*)d_in[7];
  const float* proj_b   = (const float*)d_in[8];
  const float* ph_w     = (const float*)d_in[9];
  const float* ph_b     = (const float*)d_in[10];
  const float* ln_g     = (const float*)d_in[11];
  const float* ln_b     = (const float*)d_in[12];
  float* outF = (float*)d_out;

  char* ws = (char*)d_ws;
  const size_t MB = 1024ull * 1024ull;
  float*  Abuf  = (float*) (ws + 0);            // 64MB: g2_f32 -> px_f32
  float*  Bbuf  = (float*) (ws + 64  * MB);     // 64MB: hn -> po_f32
  float2* Cc    = (float2*)(ws + 128 * MB);     // 128MB: spectrum -> pt_f32 (first 64MB)
  __bf16* Dbf   = (__bf16*)(ws + 256 * MB);     // 32MB: h0 -> xir -> xo (bf16)
  __bf16* Ebf   = (__bf16*)(ws + 288 * MB);     // 32MB: g1 -> po (bf16)
  __bf16* w1b   = (__bf16*)(ws + 320 * MB);     // 6MB
  __bf16* w2b   = (__bf16*)(ws + 327 * MB);     // 6MB
  __bf16* wpb   = (__bf16*)(ws + 334 * MB);     // 2MB
  __bf16* whb   = (__bf16*)(ws + 336 * MB);     // 2MB
  int*    counts = (int*)  (ws + 338 * MB);     // [8] threshold hits per batch
  int*    nkeep  = counts + 8;                  // [8] kept coefficients
  float*  accum  = (float*)(counts + 16);       // [2] loss accumulators

  const int gemmBlocks = (TOK / 128) * (DM / 128);   // 1024

  init_kernel<<<1, 32, 0, stream>>>(counts, accum);

  cvt_bf16_kernel<<<(3*DM*DM + 255)/256, 256, 0, stream>>>(conv1_w, w1b, 3*DM*DM);
  cvt_bf16_kernel<<<(3*DM*DM + 255)/256, 256, 0, stream>>>(conv2_w, w2b, 3*DM*DM);
  cvt_bf16_kernel<<<(DM*DM   + 255)/256, 256, 0, stream>>>(proj_w,  wpb, DM*DM);
  cvt_bf16_kernel<<<(DM*DM   + 255)/256, 256, 0, stream>>>(ph_w,    whb, DM*DM);

  embed_kernel<<<(unsigned)(TD / 256), 256, 0, stream>>>(x, tok_emb, pos_emb, Dbf);

  // conv1 + GELU  (bf16 out only)
  gemm_bf16_kernel<<<gemmBlocks, 256, 0, stream>>>(Dbf, w1b, conv1_b, nullptr, Ebf, DM, 3, 1);
  // conv2 + GELU  (f32 out for LN)
  gemm_bf16_kernel<<<gemmBlocks, 256, 0, stream>>>(Ebf, w2b, conv2_b, Abuf, nullptr, DM, 3, 1);
  // LN -> hn
  ln_kernel<<<TOK, 256, 0, stream>>>(Abuf, ln_g, ln_b, Bbuf, nullptr);
  // forward FFT per (b,d) line + complexity stats
  fft_fwd_kernel<<<BNUM * DM, 256, 0, stream>>>(Bbuf, Cc, counts);
  // per-batch kept-coefficient count
  ncalc_kernel<<<1, 32, 0, stream>>>(counts, nkeep);
  // masked inverse FFT -> xir (bf16)
  ifft_kernel<<<BNUM * DM, 256, 0, stream>>>(Cc, nkeep, Dbf);
  // proj_out = xir @ proj_w + b   (f32 + bf16)
  gemm_bf16_kernel<<<gemmBlocks, 256, 0, stream>>>(Dbf, wpb, proj_b, Bbuf, Ebf, DM, 1, 0);
  // x_ifft = LN(proj_out) -> d_out (f32) + bf16 copy
  ln_kernel<<<TOK, 256, 0, stream>>>(Bbuf, ln_g, ln_b, outF, Dbf);
  // proj_x = x_ifft @ ph_w + b ; proj_t = proj_out @ ph_w + b
  gemm_bf16_kernel<<<gemmBlocks, 256, 0, stream>>>(Dbf, whb, ph_b, Abuf, nullptr, DM, 1, 0);
  gemm_bf16_kernel<<<gemmBlocks, 256, 0, stream>>>(Ebf, whb, ph_b, (float*)Cc, nullptr, DM, 1, 0);
  // loss reductions
  loss_kernel<<<TOK, 256, 0, stream>>>(outF, Bbuf, Abuf, (float*)Cc, nkeep, accum);
  final_kernel<<<1, 1, 0, stream>>>(accum, nkeep, outF + TD);
}
